// RouteNet_52828097740868
// MI455X (gfx1250) — matrix operand order repaired and checked by reading, not compiled, not run on hardware
//
#include <hip/hip_runtime.h>

typedef __attribute__((ext_vector_type(16))) _Float16 v16h;
typedef __attribute__((ext_vector_type(8)))  float    v8f;

#define WPB 8           // waves per 256-thread block
#define LSD 32
#define MAXL 12

#define WMMA_F16(a,b,c) __builtin_amdgcn_wmma_f32_16x16x32_f16(false,(a),false,(b),(short)0,(c),false,false)

__device__ inline v8f splat8(float x){ v8f v;
#pragma unroll
  for(int i=0;i<8;++i) v[i]=x; return v; }

// v_rcp_f32-based activations: 1 v_exp + 1 v_rcp each, no IEEE div expansion.
__device__ inline float fastrcp_(float x){ return __builtin_amdgcn_rcpf(x); }
__device__ inline float sigmoidf_(float x){ return fastrcp_(1.f + __expf(-x)); }
__device__ inline float tanhf_(float x){ return 2.f*sigmoidf_(2.f*x) - 1.f; }
__device__ inline float seluf_(float x){
  const float scale = 1.0507009873554805f, alpha = 1.6732632423543772f;
  return x > 0.f ? scale*x : scale*alpha*(__expf(x)-1.f);
}
__device__ inline void wsync(){ __builtin_amdgcn_wave_barrier(); }

// ---------------------------------------------------------------------------
// Pack a row-major f32 [ktiles*32, ntiles*16] matrix into WMMA B-tiles (f16).
// Tile T = c*ntiles + j. Per lane: 16 contiguous halfs, lane = n + 16*((k/8)%2).
__global__ void pack_btiles(const float* __restrict__ src, int ncols,
                            int ktiles, int ntiles, _Float16* __restrict__ dst){
  int tid = blockIdx.x*blockDim.x + threadIdx.x;
  int total = ktiles*ntiles*32;
  if (tid >= total) return;
  int lane = tid & 31, T = tid >> 5;
  int c = T / ntiles, j = T % ntiles;
  int n = j*16 + (lane & 15);
  int kbase = c*32 + ((lane>>4)<<3);
  _Float16* d = dst + ((size_t)T*32 + lane)*16;
#pragma unroll
  for (int i=0;i<16;++i) d[i] = (_Float16)src[(size_t)(kbase + (i<8 ? i : i+8))*ncols + n];
}

// ---------------------------------------------------------------------------
__global__ void init_path_kernel(const float* __restrict__ traffic,
                                 float* __restrict__ pstate, int P){
  int i = blockIdx.x*blockDim.x + threadIdx.x;
  if (i < P*LSD) pstate[i] = ((i & 31) == 0) ? traffic[i>>5] : 0.f;
}
__global__ void init_link_kernel(const float* __restrict__ cap,
                                 float* __restrict__ lstate,
                                 float* __restrict__ macc, int NL){
  int i = blockIdx.x*blockDim.x + threadIdx.x;
  if (i < NL*LSD){ lstate[i] = ((i & 31) == 0) ? cap[i>>5] : 0.f; macc[i] = 0.f; }
}
__global__ void zero_int_kernel(int* __restrict__ p, int count){
  int i = blockIdx.x*blockDim.x + threadIdx.x;
  if (i < count) p[i] = 0;
}
__global__ void edge_meta_kernel(const int* __restrict__ paths,
                                 const int* __restrict__ seqs,
                                 int* __restrict__ offs, int* __restrict__ lens, int E){
  int e = blockIdx.x*blockDim.x + threadIdx.x;
  if (e >= E) return;
  int p = paths[e], s = seqs[e];
  if (s == 0) offs[p] = e;
  atomicMax(&lens[p], s + 1);
}

// ---------------------------------------------------------------------------
// Per-wave tile of 16 paths; sequential GRU over up to 12 hops; all gate
// matmuls are v_wmma_f32_16x16x32_f16; segment-sum via global f32 atomics.
__global__ __launch_bounds__(256) void path_gru_kernel(
    const float* __restrict__ lstate, float* __restrict__ pstate,
    float* __restrict__ macc, const int* __restrict__ links,
    const int* __restrict__ offs, const int* __restrict__ lens,
    const _Float16* __restrict__ wt, const float* __restrict__ b_path, int nPaths){
  __shared__ _Float16 sXB[WPB][16][32];
  __shared__ _Float16 sHB[WPB][16][32];
  __shared__ int      sLT[WPB][16][MAXL];
  __shared__ int      sLEN[WPB][16];
  __shared__ int      sOFF[WPB][16];

  const int lane = threadIdx.x & 31;
  const int wid  = threadIdx.x >> 5;
  const int hf   = lane >> 4;
  const int n    = lane & 15;
  const int nTiles = nPaths >> 4;

  // 12 resident weight B-tiles: K_path tiles 0..5, R_path tiles 6..11
  v16h K6[6], R6[6];
  {
    const v16h* wp = (const v16h*)wt;
#pragma unroll
    for (int i=0;i<6;++i) K6[i] = wp[(size_t)i*32 + lane];
#pragma unroll
    for (int i=0;i<6;++i) R6[i] = wp[(size_t)(6+i)*32 + lane];
  }
  const float* b0 = b_path; const float* b1 = b_path + 96;
  const float bz0 = b0[n]    + b1[n],    bz1 = b0[16+n] + b1[16+n];
  const float br0 = b0[32+n] + b1[32+n], br1 = b0[48+n] + b1[48+n];
  const float bx0 = b0[64+n], bx1 = b0[80+n];
  const float bg0 = b1[64+n], bg1 = b1[80+n];

  int wave  = blockIdx.x*(blockDim.x>>5) + wid;
  int nWave = gridDim.x*(blockDim.x>>5);
  for (int tile = wave; tile < nTiles; tile += nWave){
    const int p0 = tile << 4;
    if (lane < 16){ sLEN[wid][lane] = lens[p0+lane]; sOFF[wid][lane] = offs[p0+lane]; }
    wsync();
    for (int idx = lane; idx < 16*MAXL; idx += 32){
      int m = idx / MAXL, t = idx % MAXL;
      sLT[wid][m][t] = (t < sLEN[wid][m]) ? links[sOFF[wid][m] + t] : 0;
    }
    // path state in C-layout f32
    v8f hc0, hc1;
    {
      const float* pr = pstate + (size_t)p0*LSD;
#pragma unroll
      for (int j=0;j<8;++j){
        int m = j + (hf<<3);
        hc0[j] = pr[m*LSD + n];
        hc1[j] = pr[m*LSD + 16 + n];
      }
    }
    wsync();
    int lm[8];
#pragma unroll
    for (int j=0;j<8;++j) lm[j] = sLEN[wid][j + (hf<<3)];

    for (int t=0;t<MAXL;++t){
      // stage h (f16) for A-layout re-read
#pragma unroll
      for (int j=0;j<8;++j){
        int m = j + (hf<<3);
        sHB[wid][m][n]    = (_Float16)hc0[j];
        sHB[wid][m][16+n] = (_Float16)hc1[j];
      }
      // gather link row for hop t (lane pair per row)
      {
        int l = sLT[wid][n][t];
        const float4* src = (const float4*)(lstate + (size_t)l*LSD + hf*16);
#pragma unroll
        for (int q=0;q<4;++q){
          float4 v = src[q];
          sXB[wid][n][hf*16 + q*4 + 0] = (_Float16)v.x;
          sXB[wid][n][hf*16 + q*4 + 1] = (_Float16)v.y;
          sXB[wid][n][hf*16 + q*4 + 2] = (_Float16)v.z;
          sXB[wid][n][hf*16 + q*4 + 3] = (_Float16)v.w;
        }
        if (t+1 < MAXL){
          int ln = sLT[wid][n][t+1];
          __builtin_prefetch(lstate + (size_t)ln*LSD + hf*16, 0, 0);
        }
      }
      wsync();
      // A tiles
      v16h ax, ah;
      {
        const int kb = hf<<3;
#pragma unroll
        for (int i=0;i<8;++i){
          ax[i]   = sXB[wid][n][kb+i];
          ax[8+i] = sXB[wid][n][16+kb+i];
          ah[i]   = sHB[wid][n][kb+i];
          ah[8+i] = sHB[wid][n][16+kb+i];
        }
      }
      wsync();
      // 12 WMMAs: z, r fully fused (x@K + h@R + b), h-gate halves kept apart
      v8f cz0 = splat8(bz0), cz1 = splat8(bz1);
      v8f cr0 = splat8(br0), cr1 = splat8(br1);
      v8f cx0 = splat8(bx0), cx1 = splat8(bx1);
      v8f cg0 = splat8(bg0), cg1 = splat8(bg1);
      cz0 = WMMA_F16(ax, K6[0], cz0); cz0 = WMMA_F16(ah, R6[0], cz0);
      cz1 = WMMA_F16(ax, K6[1], cz1); cz1 = WMMA_F16(ah, R6[1], cz1);
      cr0 = WMMA_F16(ax, K6[2], cr0); cr0 = WMMA_F16(ah, R6[2], cr0);
      cr1 = WMMA_F16(ax, K6[3], cr1); cr1 = WMMA_F16(ah, R6[3], cr1);
      cx0 = WMMA_F16(ax, K6[4], cx0);
      cx1 = WMMA_F16(ax, K6[5], cx1);
      cg0 = WMMA_F16(ah, R6[4], cg0);
      cg1 = WMMA_F16(ah, R6[5], cg1);
      // gates + mask + scatter (segment_sum into links)
#pragma unroll
      for (int j=0;j<8;++j){
        float z0 = sigmoidf_(cz0[j]), z1 = sigmoidf_(cz1[j]);
        float r0 = sigmoidf_(cr0[j]), r1 = sigmoidf_(cr1[j]);
        float g0 = tanhf_(cx0[j] + r0*cg0[j]);
        float g1 = tanhf_(cx1[j] + r1*cg1[j]);
        float n0 = z0*hc0[j] + (1.f - z0)*g0;
        float n1 = z1*hc1[j] + (1.f - z1)*g1;
        if (t < lm[j]){
          hc0[j] = n0; hc1[j] = n1;
          int lrow = sLT[wid][j + (hf<<3)][t];
          atomicAdd(&macc[(size_t)lrow*LSD + n],      n0);
          atomicAdd(&macc[(size_t)lrow*LSD + 16 + n], n1);
        }
      }
      wsync();
    }
    // write back final path state
    {
      float* pr = pstate + (size_t)p0*LSD;
#pragma unroll
      for (int j=0;j<8;++j){
        int m = j + (hf<<3);
        pr[m*LSD + n]      = hc0[j];
        pr[m*LSD + 16 + n] = hc1[j];
      }
    }
  }
}

// ---------------------------------------------------------------------------
// Per-wave tile of 16 links; one GRU step with x = aggregated messages.
__global__ __launch_bounds__(256) void link_gru_kernel(
    float* __restrict__ lstate, float* __restrict__ macc,
    const _Float16* __restrict__ wt, const float* __restrict__ b_link, int nLinks){
  __shared__ _Float16 sXB[WPB][16][32];
  __shared__ _Float16 sHB[WPB][16][32];

  const int lane = threadIdx.x & 31;
  const int wid  = threadIdx.x >> 5;
  const int hf   = lane >> 4;
  const int n    = lane & 15;
  const int nTiles = nLinks >> 4;

  v16h K6[6], R6[6];
  {
    const v16h* wp = (const v16h*)wt;
#pragma unroll
    for (int i=0;i<6;++i) K6[i] = wp[(size_t)(12+i)*32 + lane];
#pragma unroll
    for (int i=0;i<6;++i) R6[i] = wp[(size_t)(18+i)*32 + lane];
  }
  const float* b0 = b_link; const float* b1 = b_link + 96;
  const float bz0 = b0[n]    + b1[n],    bz1 = b0[16+n] + b1[16+n];
  const float br0 = b0[32+n] + b1[32+n], br1 = b0[48+n] + b1[48+n];
  const float bx0 = b0[64+n], bx1 = b0[80+n];
  const float bg0 = b1[64+n], bg1 = b1[80+n];

  int wave  = blockIdx.x*(blockDim.x>>5) + wid;
  int nWave = gridDim.x*(blockDim.x>>5);
  for (int tile = wave; tile < nTiles; tile += nWave){
    const int l0 = tile << 4;
    // x = macc rows -> f16; then zero macc for next iteration
    {
      float* mr = macc + (size_t)(l0 + n)*LSD + hf*16;
#pragma unroll
      for (int q=0;q<16;++q){
        sXB[wid][n][hf*16 + q] = (_Float16)mr[q];
        mr[q] = 0.f;
      }
    }
    v8f hc0, hc1;
    {
      const float* pr = lstate + (size_t)l0*LSD;
#pragma unroll
      for (int j=0;j<8;++j){
        int m = j + (hf<<3);
        hc0[j] = pr[m*LSD + n];
        hc1[j] = pr[m*LSD + 16 + n];
      }
    }
#pragma unroll
    for (int j=0;j<8;++j){
      int m = j + (hf<<3);
      sHB[wid][m][n]    = (_Float16)hc0[j];
      sHB[wid][m][16+n] = (_Float16)hc1[j];
    }
    wsync();
    v16h ax, ah;
    {
      const int kb = hf<<3;
#pragma unroll
      for (int i=0;i<8;++i){
        ax[i]   = sXB[wid][n][kb+i];
        ax[8+i] = sXB[wid][n][16+kb+i];
        ah[i]   = sHB[wid][n][kb+i];
        ah[8+i] = sHB[wid][n][16+kb+i];
      }
    }
    wsync();
    v8f cz0 = splat8(bz0), cz1 = splat8(bz1);
    v8f cr0 = splat8(br0), cr1 = splat8(br1);
    v8f cx0 = splat8(bx0), cx1 = splat8(bx1);
    v8f cg0 = splat8(bg0), cg1 = splat8(bg1);
    cz0 = WMMA_F16(ax, K6[0], cz0); cz0 = WMMA_F16(ah, R6[0], cz0);
    cz1 = WMMA_F16(ax, K6[1], cz1); cz1 = WMMA_F16(ah, R6[1], cz1);
    cr0 = WMMA_F16(ax, K6[2], cr0); cr0 = WMMA_F16(ah, R6[2], cr0);
    cr1 = WMMA_F16(ax, K6[3], cr1); cr1 = WMMA_F16(ah, R6[3], cr1);
    cx0 = WMMA_F16(ax, K6[4], cx0);
    cx1 = WMMA_F16(ax, K6[5], cx1);
    cg0 = WMMA_F16(ah, R6[4], cg0);
    cg1 = WMMA_F16(ah, R6[5], cg1);
    {
      float* pr = lstate + (size_t)l0*LSD;
#pragma unroll
      for (int j=0;j<8;++j){
        float z0 = sigmoidf_(cz0[j]), z1 = sigmoidf_(cz1[j]);
        float r0 = sigmoidf_(cr0[j]), r1 = sigmoidf_(cr1[j]);
        float g0 = tanhf_(cx0[j] + r0*cg0[j]);
        float g1 = tanhf_(cx1[j] + r1*cg1[j]);
        int m = j + (hf<<3);
        pr[m*LSD + n]      = z0*hc0[j] + (1.f - z0)*g0;
        pr[m*LSD + 16 + n] = z1*hc1[j] + (1.f - z1)*g1;
      }
    }
    wsync();
  }
}

// ---------------------------------------------------------------------------
// Readout MLP: selu(ps@W1+b1) -> selu(@W2+b2) -> concat -> @Wf+bf.
__global__ __launch_bounds__(256) void readout_kernel(
    const float* __restrict__ pstate, const _Float16* __restrict__ wt,
    const float* __restrict__ b1, const float* __restrict__ b2,
    const float* __restrict__ Wf, const float* __restrict__ bf,
    float* __restrict__ out, int nPaths){
  __shared__ _Float16 sXB[WPB][16][32];
  __shared__ float    sPS[WPB][16][32];
  __shared__ _Float16 sHB[WPB][16][32];
  __shared__ float    sH2[WPB][16][16];

  const int lane = threadIdx.x & 31;
  const int wid  = threadIdx.x >> 5;
  const int hf   = lane >> 4;
  const int n    = lane & 15;
  const int nTiles = nPaths >> 4;
  const v16h* wp = (const v16h*)wt;   // W1 tiles at 24.., W2 tiles at 40..

  int wave  = blockIdx.x*(blockDim.x>>5) + wid;
  int nWave = gridDim.x*(blockDim.x>>5);
  for (int tile = wave; tile < nTiles; tile += nWave){
    const int p0 = tile << 4;
    {
      const float4* src = (const float4*)(pstate + (size_t)(p0 + n)*LSD + hf*16);
#pragma unroll
      for (int q=0;q<4;++q){
        float4 v = src[q];
        float vv[4] = {v.x, v.y, v.z, v.w};
#pragma unroll
        for (int r=0;r<4;++r){
          sPS[wid][n][hf*16 + q*4 + r] = vv[r];
          sXB[wid][n][hf*16 + q*4 + r] = (_Float16)vv[r];
        }
      }
    }
    wsync();
    v16h aps;
    {
      const int kb = hf<<3;
#pragma unroll
      for (int i=0;i<8;++i){
        aps[i]   = sXB[wid][n][kb+i];
        aps[8+i] = sXB[wid][n][16+kb+i];
      }
    }
    wsync();
    // h1 = selu(ps@W1+b1), converted to 8 A-tiles along K
    v16h a1[8];
#pragma unroll
    for (int c=0;c<8;++c){
#pragma unroll
      for (int sub=0; sub<2; ++sub){
        int j = 2*c + sub;
        v8f C = splat8(b1[j*16 + n]);
        v16h wb = wp[(size_t)(24 + j)*32 + lane];
        C = WMMA_F16(aps, wb, C);
#pragma unroll
        for (int jj=0;jj<8;++jj){
          int m = jj + (hf<<3);
          sHB[wid][m][sub*16 + n] = (_Float16)seluf_(C[jj]);
        }
      }
      wsync();
      {
        const int kb = hf<<3;
#pragma unroll
        for (int i=0;i<8;++i){
          a1[c][i]   = sHB[wid][n][kb+i];
          a1[c][8+i] = sHB[wid][n][16+kb+i];
        }
      }
      wsync();
    }
    // h2 n-tiles, fold directly into the 288x2 final projection.
    // Each lane owns (row m = lane&15, output o = lane>>4).
    float acc = 0.f;
    const int o = hf;
    for (int j=0;j<16;++j){
      v8f C = splat8(b2[j*16 + n]);
#pragma unroll
      for (int c=0;c<8;++c){
        v16h wb = wp[(size_t)(40 + c*16 + j)*32 + lane];
        C = WMMA_F16(a1[c], wb, C);
      }
#pragma unroll
      for (int jj=0;jj<8;++jj){
        int m = jj + (hf<<3);
        sH2[wid][m][n] = seluf_(C[jj]);
      }
      wsync();
#pragma unroll
      for (int nn=0;nn<16;++nn)
        acc += sH2[wid][n][nn] * Wf[(size_t)(j*16 + nn)*2 + o];
      wsync();
    }
#pragma unroll
    for (int nn=0;nn<32;++nn)
      acc += sPS[wid][n][nn] * Wf[(size_t)(256 + nn)*2 + o];
    out[(size_t)(p0 + n)*2 + o] = acc + bf[o];
    wsync();
  }
}

// ---------------------------------------------------------------------------
extern "C" void kernel_launch(void* const* d_in, const int* in_sizes, int n_in,
                              void* d_out, int out_size, void* d_ws, size_t ws_size,
                              hipStream_t stream){
  const float* capacities = (const float*)d_in[0];
  const float* traffic    = (const float*)d_in[1];
  const int*   links      = (const int*)d_in[2];
  const int*   paths      = (const int*)d_in[3];
  const int*   seqs       = (const int*)d_in[4];
  const float* K_link = (const float*)d_in[7];
  const float* R_link = (const float*)d_in[8];
  const float* b_link = (const float*)d_in[9];
  const float* K_path = (const float*)d_in[10];
  const float* R_path = (const float*)d_in[11];
  const float* b_path = (const float*)d_in[12];
  const float* W1 = (const float*)d_in[13];
  const float* b1 = (const float*)d_in[14];
  const float* W2 = (const float*)d_in[15];
  const float* b2 = (const float*)d_in[16];
  const float* Wf = (const float*)d_in[17];
  const float* bf = (const float*)d_in[18];

  const int NL = in_sizes[0];       // 10000
  const int P  = in_sizes[1];       // 100000
  const int E  = in_sizes[2];       // ~800000

  // ws carve (256B aligned regions)
  char* w = (char*)d_ws;
  auto carve = [&](size_t bytes)->void*{
    void* p = (void*)w; w += (bytes + 255) & ~(size_t)255; return p; };
  _Float16* wt     = (_Float16*)carve((size_t)168*512*sizeof(_Float16)); // 168 B-tiles
  float*    pstate = (float*)carve((size_t)P*LSD*sizeof(float));
  float*    lstate = (float*)carve((size_t)NL*LSD*sizeof(float));
  float*    macc   = (float*)carve((size_t)NL*LSD*sizeof(float));
  int*      offs   = (int*)carve((size_t)P*sizeof(int));
  int*      lens   = (int*)carve((size_t)P*sizeof(int));

  // --- weight packing into WMMA B-tile layout (f16) ---
  // tiles: K_path[0..5], R_path[6..11], K_link[12..17], R_link[18..23],
  //        W1[24..39], W2[40..167]
  pack_btiles<<<1, 256, 0, stream>>>(K_path, 96, 1, 6, wt + (size_t)0*512);
  pack_btiles<<<1, 256, 0, stream>>>(R_path, 96, 1, 6, wt + (size_t)6*512);
  pack_btiles<<<1, 256, 0, stream>>>(K_link, 96, 1, 6, wt + (size_t)12*512);
  pack_btiles<<<1, 256, 0, stream>>>(R_link, 96, 1, 6, wt + (size_t)18*512);
  pack_btiles<<<2, 256, 0, stream>>>(W1, 256, 1, 16, wt + (size_t)24*512);
  pack_btiles<<<16, 256, 0, stream>>>(W2, 256, 8, 16, wt + (size_t)40*512);

  // --- state / meta init (re-done every call: deterministic) ---
  init_path_kernel<<<(P*LSD + 255)/256, 256, 0, stream>>>(traffic, pstate, P);
  init_link_kernel<<<(NL*LSD + 255)/256, 256, 0, stream>>>(capacities, lstate, macc, NL);
  zero_int_kernel<<<(P + 255)/256, 256, 0, stream>>>(lens, P);
  edge_meta_kernel<<<(E + 255)/256, 256, 0, stream>>>(paths, seqs, offs, lens, E);

  // --- T = 4 message-passing iterations ---
  const int pathBlocks = ((P >> 4) + WPB - 1) / WPB;   // 6250 tiles -> 782 blocks
  const int linkBlocks = ((NL >> 4) + WPB - 1) / WPB;  // 625 tiles  -> 79 blocks
  for (int it = 0; it < 4; ++it){
    path_gru_kernel<<<pathBlocks, 256, 0, stream>>>(
        lstate, pstate, macc, links, offs, lens, wt, b_path, P);
    link_gru_kernel<<<linkBlocks, 256, 0, stream>>>(
        lstate, macc, wt, b_link, NL);
  }

  // --- readout MLP ---
  readout_kernel<<<pathBlocks, 256, 0, stream>>>(
      pstate, wt, b1, b2, Wf, bf, (float*)d_out, P);
}
